// AttnBlock_15899968930148
// MI455X (gfx1250) — matrix operand order repaired and checked by reading, not compiled, not run on hardware
//
#include <hip/hip_runtime.h>
#include <hip/hip_bf16.h>

// ---------------------------------------------------------------------------
// AttnBlock (GroupNorm -> QKV -> softmax(QK^T)V -> proj + residual)
// B=4, C=512, H=W=64 (N=4096). fp32 in/out, bf16 WMMA internally.
//
// Workspace layout (bf16/ushort elements unless noted):
//   hn  (B,N,C)  normalized input, transposed       8,388,608
//   qt  (B,N,C)  Q * C^-0.5                         8,388,608
//   kt  (B,N,C)  K                                  8,388,608
//   vv  (B,C,N)  V (kept channel-major for PV B-frags) 8,388,608
//   h2  (B,N,C)  attention output                   8,388,608
//   wqb/wkb/wvb/wpb (C,C) bf16 weights              4 * 262,144
//   stats: 128*2 f32 (mean, rstd)
// Total ~86 MB.
// ---------------------------------------------------------------------------

#define BATCH 4
#define CCH   512
#define NTOK  4096
#define NGRP  32
#define GCH   16      // channels per group
#define EPSV  1e-6f

typedef __attribute__((ext_vector_type(16))) __bf16          v16bf;
typedef __attribute__((ext_vector_type(8)))  float           v8f;
typedef __attribute__((ext_vector_type(8)))  unsigned short  v8u;

// A-fragment: 16x32 bf16, per lane two 16-byte chunks:
// lane<16: K = {0..15}; lane>=16: K = {16..31}
union AFrag { v16bf bf; struct { v8u lo; v8u hi; } p; };

__device__ __forceinline__ unsigned short f2bf(float f) {
    union { float f; unsigned u; } x; x.f = f;
    unsigned r = (x.u + 0x7FFFu + ((x.u >> 16) & 1u)) >> 16;
    return (unsigned short)r;
}

__device__ __forceinline__ v8f wmma_bf16(v16bf a, v16bf b, v8f c) {
    // D = A(16x32) * B(32x16) + C, f32 accum
    return __builtin_amdgcn_wmma_f32_16x16x32_bf16(
        /*neg_a=*/false, a, /*neg_b=*/false, b,
        /*c_mod=*/(short)0, c, /*reuse_a=*/false, /*reuse_b=*/false);
}

// ---------------------------------------------------------------------------
// 1) GroupNorm statistics: one block per (batch, group)
// ---------------------------------------------------------------------------
__global__ __launch_bounds__(256) void gn_stats_kernel(
        const float* __restrict__ x, float* __restrict__ stats) {
    const int bg = blockIdx.x;                       // b*32+g ; channels contiguous
    const float* p = x + (size_t)bg * (GCH * NTOK);
    float s = 0.f, q = 0.f;
    for (int i = threadIdx.x; i < GCH * NTOK; i += 256) {
        float v = p[i]; s += v; q += v * v;
    }
    __shared__ float ss[256], sq[256];
    ss[threadIdx.x] = s; sq[threadIdx.x] = q;
    __syncthreads();
    for (int st = 128; st > 0; st >>= 1) {
        if (threadIdx.x < st) {
            ss[threadIdx.x] += ss[threadIdx.x + st];
            sq[threadIdx.x] += sq[threadIdx.x + st];
        }
        __syncthreads();
    }
    if (threadIdx.x == 0) {
        const float inv = 1.0f / (float)(GCH * NTOK);
        float mu  = ss[0] * inv;
        float var = sq[0] * inv - mu * mu;
        stats[2 * bg]     = mu;
        stats[2 * bg + 1] = rsqrtf(var + EPSV);
    }
}

// ---------------------------------------------------------------------------
// 2) GroupNorm apply + transpose to (B,N,C) bf16
// ---------------------------------------------------------------------------
__global__ __launch_bounds__(256) void gn_apply_kernel(
        const float* __restrict__ x, const float* __restrict__ gw,
        const float* __restrict__ gb, const float* __restrict__ stats,
        unsigned short* __restrict__ hn) {
    int e = blockIdx.x * 256 + threadIdx.x;          // < B*C*N
    int b = e / (CCH * NTOK);
    int r = e % (CCH * NTOK);
    int c = r / NTOK, n = r % NTOK;
    int bg = b * NGRP + (c / GCH);
    float mu = stats[2 * bg], rs = stats[2 * bg + 1];
    float v = (x[e] - mu) * rs * gw[c] + gb[c];
    hn[(size_t)b * NTOK * CCH + (size_t)n * CCH + c] = f2bf(v);
}

// ---------------------------------------------------------------------------
// 3) Convert weights fp32 -> bf16
// ---------------------------------------------------------------------------
__global__ __launch_bounds__(256) void wconv_kernel(
        const float* __restrict__ wq, const float* __restrict__ wk,
        const float* __restrict__ wv, const float* __restrict__ wp,
        unsigned short* __restrict__ oq, unsigned short* __restrict__ ok,
        unsigned short* __restrict__ ov, unsigned short* __restrict__ op) {
    int i = blockIdx.x * 256 + threadIdx.x;          // < C*C
    oq[i] = f2bf(wq[i]);
    ok[i] = f2bf(wk[i]);
    ov[i] = f2bf(wv[i]);
    op[i] = f2bf(wp[i]);
}

// ---------------------------------------------------------------------------
// 4) Fused QKV GEMM: (N,C) x (C,C)^T per batch, one wave per 16x16 tile,
//    A-fragment reused across the three weight matrices.
//    qt/kt stored (N,C); v stored (C,N) (packed 16B stores).
// ---------------------------------------------------------------------------
__global__ __launch_bounds__(128) void qkv_kernel(
        const unsigned short* __restrict__ hn,
        const unsigned short* __restrict__ wqb, const unsigned short* __restrict__ wkb,
        const unsigned short* __restrict__ wvb,
        const float* __restrict__ bq, const float* __restrict__ bk,
        const float* __restrict__ bv,
        unsigned short* __restrict__ qt, unsigned short* __restrict__ kt,
        unsigned short* __restrict__ vv) {
    const int wave = threadIdx.x >> 5, lane = threadIdx.x & 31;
    const int half = lane >> 4, lm = lane & 15;
    int tile = blockIdx.x * 4 + wave;                 // < B * 256 * 32
    int b  = tile / (256 * 32);
    int t  = tile % (256 * 32);
    int n0 = (t >> 5) * 16;
    int o0 = (t & 31) * 16;

    const unsigned short* Arow = hn  + (size_t)b * NTOK * CCH + (size_t)(n0 + lm) * CCH;
    const unsigned short* Bq   = wqb + (size_t)(o0 + lm) * CCH;
    const unsigned short* Bk   = wkb + (size_t)(o0 + lm) * CCH;
    const unsigned short* Bv   = wvb + (size_t)(o0 + lm) * CCH;

    v8f aq = {}, ak = {}, av = {};
#pragma unroll
    for (int kk = 0; kk < CCH; kk += 32) {
        AFrag a;
        a.p.lo = *(const v8u*)(Arow + kk + half * 16);
        a.p.hi = *(const v8u*)(Arow + kk + half * 16 + 8);
        v16bf fq = *(const v16bf*)(Bq + kk + half * 16);
        v16bf fk = *(const v16bf*)(Bk + kk + half * 16);
        v16bf fv = *(const v16bf*)(Bv + kk + half * 16);
        aq = wmma_bf16(a.bf, fq, aq);
        ak = wmma_bf16(a.bf, fk, ak);
        av = wmma_bf16(a.bf, fv, av);
    }

    const float biasq = bq[o0 + lm], biask = bk[o0 + lm], biasv = bv[o0 + lm];
    const float qscale = 0.044194173824159216f;       // 512^-0.5 baked into Q

    const size_t boff = (size_t)b * NTOK * CCH;
#pragma unroll
    for (int r = 0; r < 8; r++) {
        int n = n0 + r + 8 * half;
        qt[boff + (size_t)n * CCH + o0 + lm] = f2bf((aq[r] + biasq) * qscale);
        kt[boff + (size_t)n * CCH + o0 + lm] = f2bf(ak[r] + biask);
    }
    v8u pv;
#pragma unroll
    for (int r = 0; r < 8; r++) pv[r] = f2bf(av[r] + biasv);
    *(v8u*)(vv + (size_t)b * CCH * NTOK + (size_t)(o0 + lm) * NTOK + n0 + 8 * half) = pv;
}

// ---------------------------------------------------------------------------
// 5) Flash attention. Grid (N/16, B), 128 threads = 4 waves.
//    Each WG: 16 queries. j-blocks of 64 keys; wave w computes S columns
//    [w*16, w*16+16) (16 WMMAs, K=C=512) and owns output channels
//    [w*128, w*128+128) (8 f32 accum tiles, 16 PV WMMAs per block).
//    Online softmax shared via LDS (2 barriers / iteration).
// ---------------------------------------------------------------------------
__global__ __launch_bounds__(128) void attn_kernel(
        const unsigned short* __restrict__ qt, const unsigned short* __restrict__ kt,
        const unsigned short* __restrict__ vv, unsigned short* __restrict__ h2) {
    const int wave = threadIdx.x >> 5, lane = threadIdx.x & 31;
    const int half = lane >> 4, lm = lane & 15;
    const int b  = blockIdx.y;
    const int q0 = blockIdx.x * 16;
    const size_t nc = (size_t)NTOK * CCH;
    const unsigned short* Q = qt + (size_t)b * nc;
    const unsigned short* K = kt + (size_t)b * nc;
    const unsigned short* V = vv + (size_t)b * nc;
    const int c0 = wave * 128;                        // this wave's output channels

    __shared__ unsigned short ptile[16][72];          // 16 rows x 64 cols (+8 pad)
    __shared__ float lmax[4][16];
    __shared__ float lsum[4][16];

    v8f acc[8] = {};
    float mo[8], li[8];
#pragma unroll
    for (int r = 0; r < 8; r++) { mo[r] = -3.0e38f; li[r] = 0.f; }

    const unsigned short* Arow = Q + (size_t)(q0 + lm) * CCH;

    for (int jb = 0; jb < NTOK; jb += 64) {
        if (jb + 64 < NTOK) {                          // stream next K/V block
            __builtin_prefetch(K + (size_t)(jb + 64 + wave * 16 + lm) * CCH, 0, 1);
            __builtin_prefetch(V + (size_t)(c0 + lm) * NTOK + jb + 64, 0, 1);
        }

        // ---- S tile: rows q0..q0+15, cols jw..jw+15 ----
        const int jw = jb + wave * 16;
        const unsigned short* Brow = K + (size_t)(jw + lm) * CCH;
        v8f s = {};
#pragma unroll
        for (int kk = 0; kk < CCH; kk += 32) {
            AFrag a;
            a.p.lo = *(const v8u*)(Arow + kk + half * 16);
            a.p.hi = *(const v8u*)(Arow + kk + half * 16 + 8);
            v16bf bf = *(const v16bf*)(Brow + kk + half * 16);
            s = wmma_bf16(a.bf, bf, s);
        }

        // ---- local row-max (reduce over 16 lanes of this half) ----
        float rm[8];
#pragma unroll
        for (int r = 0; r < 8; r++) {
            float t = s[r];
            t = fmaxf(t, __shfl_xor(t, 1, 32));
            t = fmaxf(t, __shfl_xor(t, 2, 32));
            t = fmaxf(t, __shfl_xor(t, 4, 32));
            t = fmaxf(t, __shfl_xor(t, 8, 32));
            rm[r] = t;
        }
        if (lm == 0) {
#pragma unroll
            for (int r = 0; r < 8; r++) lmax[wave][r + 8 * half] = rm[r];
        }
        __syncthreads();

        // ---- combine maxes, p = exp(s - m_new), row sums, write P tile ----
        float mn[8], al[8];
#pragma unroll
        for (int r = 0; r < 8; r++) {
            int row = r + 8 * half;
            float m = mo[r];
            m = fmaxf(m, lmax[0][row]); m = fmaxf(m, lmax[1][row]);
            m = fmaxf(m, lmax[2][row]); m = fmaxf(m, lmax[3][row]);
            mn[r] = m;
            al[r] = __expf(mo[r] - m);
            mo[r] = m;
        }
        float rs[8];
#pragma unroll
        for (int r = 0; r < 8; r++) {
            float p = __expf(s[r] - mn[r]);
            float t = p;
            t += __shfl_xor(t, 1, 32);
            t += __shfl_xor(t, 2, 32);
            t += __shfl_xor(t, 4, 32);
            t += __shfl_xor(t, 8, 32);
            rs[r] = t;
            ptile[r + 8 * half][wave * 16 + lm] = f2bf(p);
        }
        if (lm == 0) {
#pragma unroll
            for (int r = 0; r < 8; r++) lsum[wave][r + 8 * half] = rs[r];
        }
        __syncthreads();

        // ---- l update, rescale accumulators, PV WMMAs ----
#pragma unroll
        for (int r = 0; r < 8; r++) {
            int row = r + 8 * half;
            float add = lsum[0][row] + lsum[1][row] + lsum[2][row] + lsum[3][row];
            li[r] = li[r] * al[r] + add;
        }
#pragma unroll
        for (int t = 0; t < 8; t++)
#pragma unroll
            for (int r = 0; r < 8; r++) acc[t][r] *= al[r];

        AFrag pa0, pa1;                                // P as A-frags (K=j)
        {
            const unsigned short* pr = &ptile[lm][0];
            pa0.p.lo = *(const v8u*)(pr + half * 16);
            pa0.p.hi = *(const v8u*)(pr + half * 16 + 8);
            pa1.p.lo = *(const v8u*)(pr + 32 + half * 16);
            pa1.p.hi = *(const v8u*)(pr + 32 + half * 16 + 8);
        }
#pragma unroll
        for (int t = 0; t < 8; t++) {
            const unsigned short* Vrow = V + (size_t)(c0 + t * 16 + lm) * NTOK + jb;
            v16bf b0 = *(const v16bf*)(Vrow + half * 16);
            v16bf b1 = *(const v16bf*)(Vrow + 32 + half * 16);
            acc[t] = wmma_bf16(pa0.bf, b0, acc[t]);
            acc[t] = wmma_bf16(pa1.bf, b1, acc[t]);
        }
        // next iteration's LDS writes are fenced by the two barriers above
    }

    // ---- finalize: divide by l, store h2 (B,N,C) bf16 ----
#pragma unroll
    for (int r = 0; r < 8; r++) li[r] = 1.0f / li[r];
    unsigned short* H = h2 + (size_t)b * nc;
#pragma unroll
    for (int t = 0; t < 8; t++) {
        int c = c0 + t * 16 + lm;
#pragma unroll
        for (int r = 0; r < 8; r++) {
            int n = q0 + r + 8 * half;
            H[(size_t)n * CCH + c] = f2bf(acc[t][r] * li[r]);
        }
    }
}

// ---------------------------------------------------------------------------
// 6) Projection GEMM + bias + residual: out = x + h2 @ wp^T + bp, fp32 out.
// ---------------------------------------------------------------------------
__global__ __launch_bounds__(128) void proj_kernel(
        const unsigned short* __restrict__ h2, const unsigned short* __restrict__ wpb,
        const float* __restrict__ bp, const float* __restrict__ x,
        float* __restrict__ out) {
    const int wave = threadIdx.x >> 5, lane = threadIdx.x & 31;
    const int half = lane >> 4, lm = lane & 15;
    int tile = blockIdx.x * 4 + wave;
    int b  = tile / (256 * 32);
    int t  = tile % (256 * 32);
    int n0 = (t >> 5) * 16;
    int o0 = (t & 31) * 16;

    const unsigned short* Arow = h2  + (size_t)b * NTOK * CCH + (size_t)(n0 + lm) * CCH;
    const unsigned short* Bw   = wpb + (size_t)(o0 + lm) * CCH;

    v8f acc = {};
#pragma unroll
    for (int kk = 0; kk < CCH; kk += 32) {
        AFrag a;
        a.p.lo = *(const v8u*)(Arow + kk + half * 16);
        a.p.hi = *(const v8u*)(Arow + kk + half * 16 + 8);
        v16bf bf = *(const v16bf*)(Bw + kk + half * 16);
        acc = wmma_bf16(a.bf, bf, acc);
    }

    const float bias = bp[o0 + lm];
    const size_t base = (size_t)b * CCH * NTOK + (size_t)(o0 + lm) * NTOK + n0 + 8 * half;
    float4 x0 = *(const float4*)(x + base);
    float4 x1 = *(const float4*)(x + base + 4);
    float4 r0, r1;
    r0.x = acc[0] + bias + x0.x; r0.y = acc[1] + bias + x0.y;
    r0.z = acc[2] + bias + x0.z; r0.w = acc[3] + bias + x0.w;
    r1.x = acc[4] + bias + x1.x; r1.y = acc[5] + bias + x1.y;
    r1.z = acc[6] + bias + x1.z; r1.w = acc[7] + bias + x1.w;
    *(float4*)(out + base)     = r0;
    *(float4*)(out + base + 4) = r1;
}

// ---------------------------------------------------------------------------
extern "C" void kernel_launch(void* const* d_in, const int* in_sizes, int n_in,
                              void* d_out, int out_size, void* d_ws, size_t ws_size,
                              hipStream_t stream) {
    (void)in_sizes; (void)n_in; (void)out_size; (void)ws_size;
    const float* x  = (const float*)d_in[0];
    const float* gw = (const float*)d_in[1];
    const float* gb = (const float*)d_in[2];
    const float* wq = (const float*)d_in[3];
    const float* bq = (const float*)d_in[4];
    const float* wk = (const float*)d_in[5];
    const float* bk = (const float*)d_in[6];
    const float* wv = (const float*)d_in[7];
    const float* bv = (const float*)d_in[8];
    const float* wp = (const float*)d_in[9];
    const float* bp = (const float*)d_in[10];
    float* out = (float*)d_out;

    const size_t NC = (size_t)BATCH * NTOK * CCH;     // 8,388,608 elements
    unsigned short* hn  = (unsigned short*)d_ws;
    unsigned short* qt  = hn  + NC;
    unsigned short* kt  = qt  + NC;
    unsigned short* vv  = kt  + NC;
    unsigned short* h2  = vv  + NC;
    unsigned short* wqb = h2  + NC;
    unsigned short* wkb = wqb + (size_t)CCH * CCH;
    unsigned short* wvb = wkb + (size_t)CCH * CCH;
    unsigned short* wpb = wvb + (size_t)CCH * CCH;
    float* stats = (float*)(wpb + (size_t)CCH * CCH);

    wconv_kernel<<<(CCH * CCH) / 256, 256, 0, stream>>>(wq, wk, wv, wp,
                                                        wqb, wkb, wvb, wpb);
    gn_stats_kernel<<<BATCH * NGRP, 256, 0, stream>>>(x, stats);
    gn_apply_kernel<<<(BATCH * CCH * NTOK) / 256, 256, 0, stream>>>(x, gw, gb, stats, hn);

    const int tiles = BATCH * (NTOK / 16) * (CCH / 16);   // 32768
    qkv_kernel<<<tiles / 4, 128, 0, stream>>>(hn, wqb, wkb, wvb, bq, bk, bv,
                                              qt, kt, vv);
    attn_kernel<<<dim3(NTOK / 16, BATCH), 128, 0, stream>>>(qt, kt, vv, h2);
    proj_kernel<<<tiles / 4, 128, 0, stream>>>(h2, wpb, bp, x, out);
}
